// Glm4MoeMoE_41781441856071
// MI455X (gfx1250) — matrix operand order repaired and checked by reading, not compiled, not run on hardware
//
#include <hip/hip_runtime.h>
#include <hip/hip_bf16.h>

// ---------------- problem constants (from reference) ----------------
#define HDIM 2048
#define FDIM 1024
#define NEXP 16
#define NEXP1 17       // + shared pseudo-expert
#define KTOP 4
#define TMAX 2048      // power of two (== 1<<11), used for entry packing

typedef __attribute__((ext_vector_type(16))) __bf16 v16bf;
typedef __attribute__((ext_vector_type(8)))  __bf16 v8bf;
typedef __attribute__((ext_vector_type(8)))  float  v8f;

union frag16 { v16bf v; v8bf h[2]; };

// ---------------- fragment loaders -----------------------------------------
// A (16x32 bf16, M = lane&15, K-half = lane>>4):
//   elems j=0..7 -> K=kk+half*8+j ; j=8..15 -> K=kk+16+half*8+(j-8)
__device__ __forceinline__ v16bf load_frag_A(const float* __restrict__ row,
                                             int kk, int half) {
    v16bf a;
    const float* p0 = row + kk + half * 8;
    const float* p1 = row + kk + 16 + half * 8;
    #pragma unroll
    for (int j = 0; j < 8; ++j) {
        a[j]     = (__bf16)p0[j];
        a[j + 8] = (__bf16)p1[j];
    }
    return a;
}
__device__ __forceinline__ v16bf load_frag_A(const __bf16* __restrict__ row,
                                             int kk, int half) {
    frag16 f;
    f.h[0] = *(const v8bf*)(row + kk + half * 8);
    f.h[1] = *(const v8bf*)(row + kk + 16 + half * 8);
    return f.v;
}
// B (32x16, N = lane&15 = weight row, K = kk+half*8+j for j<8, kk+16+half*8+(j-8) for j>=8)
__device__ __forceinline__ v16bf load_frag_B(const float* __restrict__ row,
                                             int kk, int half) {
    v16bf b;
    const float* p0 = row + kk + half * 8;
    const float* p1 = row + kk + 16 + half * 8;
    #pragma unroll
    for (int j = 0; j < 8; ++j) {
        b[j]     = (__bf16)p0[j];
        b[j + 8] = (__bf16)p1[j];
    }
    return b;
}
__device__ __forceinline__ v16bf load_frag_B(const __bf16* __restrict__ row,
                                             int kk, int half) {
    frag16 f;
    f.h[0] = *(const v8bf*)(row + kk + half * 8);
    f.h[1] = *(const v8bf*)(row + kk + 16 + half * 8);
    return f.v;
}

__device__ __forceinline__ float fast_silu(float g) {
    return g * __builtin_amdgcn_rcpf(1.0f + __expf(-g));   // g * sigmoid(g)
}

// ---------------- fp32 -> bf16 streaming conversion -------------------------
__global__ __launch_bounds__(256) void cvt_bf16_kernel(
    const float* __restrict__ src, __bf16* __restrict__ dst, long n)
{
    long i = ((long)blockIdx.x * 256 + threadIdx.x) * 8;
    if (i + 8 <= n) {
        const float4 a = *(const float4*)(src + i);
        const float4 b = *(const float4*)(src + i + 4);
        v8bf o;
        o[0] = (__bf16)a.x; o[1] = (__bf16)a.y; o[2] = (__bf16)a.z; o[3] = (__bf16)a.w;
        o[4] = (__bf16)b.x; o[5] = (__bf16)b.y; o[6] = (__bf16)b.z; o[7] = (__bf16)b.w;
        *(v8bf*)(dst + i) = o;
    } else {
        for (long j = i; j < n; ++j) dst[j] = (__bf16)src[j];
    }
}

// ---------------- init: zero output, reset counts, install shared expert ----
__global__ __launch_bounds__(256) void moe_init_kernel(
    float* __restrict__ out, int out_n,
    int* __restrict__ counts,
    int* __restrict__ bucket_tok, float* __restrict__ bucket_w, int T)
{
    int i = blockIdx.x * blockDim.x + threadIdx.x;
    if (i < out_n) out[i] = 0.0f;
    if (i < NEXP1) counts[i] = (i == NEXP) ? T : 0;
    if (i < T) {
        bucket_tok[NEXP * TMAX + i] = i;   // shared expert: identity token list
        bucket_w  [NEXP * TMAX + i] = 1.0f;
    }
}

// ---------------- router: sigmoid + bias-corrected top-4, scatter ----------
__global__ __launch_bounds__(128) void router_kernel(
    const float* __restrict__ x,        // [T,H]
    const float* __restrict__ rw,       // [E,H]
    const float* __restrict__ rb,       // [E]
    int* __restrict__ counts,
    int* __restrict__ bucket_tok, float* __restrict__ bucket_w,
    int* __restrict__ tok_entry,        // [T,KTOP] packed e*TMAX+pos
    int T)
{
    int t    = blockIdx.x * 4 + (threadIdx.x >> 5);
    int lane = threadIdx.x & 31;
    if (t >= T) return;
    const float* hrow = x + (size_t)t * HDIM;

    float logit[NEXP];
    #pragma unroll 1
    for (int e = 0; e < NEXP; ++e) {
        const float* wrow = rw + (size_t)e * HDIM;
        float acc = 0.0f;
        for (int i = lane; i < HDIM; i += 32)
            acc += hrow[i] * wrow[i];
        #pragma unroll
        for (int off = 16; off > 0; off >>= 1)
            acc += __shfl_down(acc, off, 32);
        logit[e] = acc;   // valid in lane 0
    }
    if (lane == 0) {
        float score[NEXP], choice[NEXP];
        #pragma unroll
        for (int e = 0; e < NEXP; ++e) {
            float s = 1.0f / (1.0f + __expf(-logit[e]));
            score[e]  = s;
            choice[e] = s + rb[e];
        }
        int   idx[KTOP];
        float wv[KTOP];
        float wsum = 0.0f;
        #pragma unroll
        for (int k = 0; k < KTOP; ++k) {
            int best = 0; float bv = -3.0e38f;
            #pragma unroll
            for (int e = 0; e < NEXP; ++e)
                if (choice[e] > bv) { bv = choice[e]; best = e; }
            idx[k] = best; wv[k] = score[best]; wsum += score[best];
            choice[best] = -3.0e38f;
        }
        float inv = 1.0f / (wsum + 1e-20f);   // ROUTED_SCALE == 1.0
        #pragma unroll
        for (int k = 0; k < KTOP; ++k) {
            int e   = idx[k];
            int pos = atomicAdd(&counts[e], 1);
            bucket_tok[e * TMAX + pos] = t;
            bucket_w  [e * TMAX + pos] = wv[k] * inv;
            tok_entry [t * KTOP + k]   = e * TMAX + pos;
        }
    }
}

// ---------------- tiny exclusive scan over 17 expert counts ----------------
__global__ void offsets_kernel(const int* __restrict__ counts, int* __restrict__ offsets)
{
    if (threadIdx.x == 0 && blockIdx.x == 0) {
        int s = 0;
        for (int e = 0; e < NEXP1; ++e) { offsets[e] = s; s += counts[e]; }
    }
}

// ---------------- gate/up GEMM + SiLU --------------------------------------
// Wave tile: M=64 (4 A frags) x N=16, two B matrices -> 8 WMMA per K-step.
// Block: 4 waves (128 thr) covering 64 F columns. Grid: (F/64, ceil(T/64), 17)
template <typename WT>
__global__ __launch_bounds__(128) void gateup_kernel(
    const WT* __restrict__ x,         // [T,H]
    const WT* __restrict__ eg,        // [E,F,H]
    const WT* __restrict__ eu,        // [E,F,H]
    const WT* __restrict__ sg,        // [F,H]
    const WT* __restrict__ su,        // [F,H]
    const int* __restrict__ counts, const int* __restrict__ offsets,
    const int* __restrict__ bucket_tok,
    __bf16* __restrict__ act)         // [rows,F] bf16
{
    int e   = blockIdx.z;
    int m0  = blockIdx.y * 64;
    int cnt = counts[e];
    if (m0 >= cnt) return;                       // block-uniform exit

    int wave = threadIdx.x >> 5;
    int lane = threadIdx.x & 31;
    int f0   = blockIdx.x * 64 + wave * 16;

    const WT* gW = (e < NEXP) ? (eg + (size_t)e * FDIM * HDIM) : sg;
    const WT* uW = (e < NEXP) ? (eu + (size_t)e * FDIM * HDIM) : su;

    int mrow = lane & 15;
    int half = lane >> 4;

    const WT* arow[4];
    #pragma unroll
    for (int i = 0; i < 4; ++i) {
        int m   = m0 + 16 * i + mrow;
        int mcl = (m < cnt) ? m : (cnt - 1);     // clamp: keep EXEC full
        arow[i] = x + (size_t)bucket_tok[e * TMAX + mcl] * HDIM;
    }
    const WT* grow = gW + (size_t)(f0 + mrow) * HDIM;
    const WT* urow = uW + (size_t)(f0 + mrow) * HDIM;

    v8f accg[4] = {};
    v8f accu[4] = {};
    #pragma unroll 1
    for (int kk = 0; kk < HDIM; kk += 32) {
        v16bf bg = load_frag_B(grow, kk, half);
        v16bf bu = load_frag_B(urow, kk, half);
        #pragma unroll
        for (int i = 0; i < 4; ++i) {
            v16bf a = load_frag_A(arow[i], kk, half);
            accg[i] = __builtin_amdgcn_wmma_f32_16x16x32_bf16(false, a, false, bg,
                                                              (short)0, accg[i], false, false);
            accu[i] = __builtin_amdgcn_wmma_f32_16x16x32_bf16(false, a, false, bu,
                                                              (short)0, accu[i], false, false);
        }
    }

    // D layout: lane holds rows m = r + 8*half (per 16-row subtile), col n=lane&15
    int n    = lane & 15;
    int base = offsets[e];
    if (m0 + 64 <= cnt) {                         // full tile: no predication
        #pragma unroll
        for (int i = 0; i < 4; ++i) {
            #pragma unroll
            for (int r = 0; r < 8; ++r) {
                int m = m0 + 16 * i + r + 8 * half;
                act[(size_t)(base + m) * FDIM + f0 + n] =
                    (__bf16)(fast_silu(accg[i][r]) * accu[i][r]);
            }
        }
    } else {
        #pragma unroll
        for (int i = 0; i < 4; ++i) {
            #pragma unroll
            for (int r = 0; r < 8; ++r) {
                int m = m0 + 16 * i + r + 8 * half;
                if (m < cnt)
                    act[(size_t)(base + m) * FDIM + f0 + n] =
                        (__bf16)(fast_silu(accg[i][r]) * accu[i][r]);
            }
        }
    }
}

// ---------------- down GEMM: weighted partial rows (or atomic fallback) ----
// Wave tile: M=64 x N=32 (2 B frags) -> 8 WMMA per K-step.
// Block: 4 waves covering 128 H columns. Grid: (H/128, ceil(T/64), 17)
template <typename WT, bool PART>
__global__ __launch_bounds__(128) void down_kernel(
    const __bf16* __restrict__ act,   // [rows,F] bf16
    const WT* __restrict__ ed,        // [E,H,F]
    const WT* __restrict__ sd,        // [H,F]
    const int* __restrict__ counts, const int* __restrict__ offsets,
    const int* __restrict__ bucket_tok, const float* __restrict__ bucket_w,
    float* __restrict__ part,         // [rows,H] weighted partials (PART path)
    float* __restrict__ out)          // [T,H]   (atomic fallback path)
{
    int e   = blockIdx.z;
    int m0  = blockIdx.y * 64;
    int cnt = counts[e];
    if (m0 >= cnt) return;

    int wave = threadIdx.x >> 5;
    int lane = threadIdx.x & 31;
    int h0   = blockIdx.x * 128 + wave * 32;

    const WT* dW = (e < NEXP) ? (ed + (size_t)e * HDIM * FDIM) : sd;

    int mrow = lane & 15;
    int half = lane >> 4;
    int base = offsets[e];

    const __bf16* arow[4];
    #pragma unroll
    for (int i = 0; i < 4; ++i) {
        int m   = m0 + 16 * i + mrow;
        int mcl = (m < cnt) ? m : (cnt - 1);
        arow[i] = act + (size_t)(base + mcl) * FDIM;
    }
    const WT* brow0 = dW + (size_t)(h0 + mrow) * FDIM;
    const WT* brow1 = dW + (size_t)(h0 + 16 + mrow) * FDIM;

    v8f acc0[4] = {};
    v8f acc1[4] = {};
    #pragma unroll 1
    for (int kk = 0; kk < FDIM; kk += 32) {
        v16bf b0 = load_frag_B(brow0, kk, half);
        v16bf b1 = load_frag_B(brow1, kk, half);
        #pragma unroll
        for (int i = 0; i < 4; ++i) {
            v16bf a = load_frag_A(arow[i], kk, half);
            acc0[i] = __builtin_amdgcn_wmma_f32_16x16x32_bf16(false, a, false, b0,
                                                              (short)0, acc0[i], false, false);
            acc1[i] = __builtin_amdgcn_wmma_f32_16x16x32_bf16(false, a, false, b1,
                                                              (short)0, acc1[i], false, false);
        }
    }

    int n = lane & 15;
    if (PART) {
        if (m0 + 64 <= cnt) {                     // full tile: no predication
            #pragma unroll
            for (int i = 0; i < 4; ++i) {
                #pragma unroll
                for (int r = 0; r < 8; ++r) {
                    int   m = m0 + 16 * i + r + 8 * half;
                    float w = bucket_w[e * TMAX + m];
                    float* prow = part + (size_t)(base + m) * HDIM + h0;
                    prow[n]      = w * acc0[i][r];
                    prow[16 + n] = w * acc1[i][r];
                }
            }
        } else {
            #pragma unroll
            for (int i = 0; i < 4; ++i) {
                #pragma unroll
                for (int r = 0; r < 8; ++r) {
                    int m = m0 + 16 * i + r + 8 * half;
                    if (m < cnt) {
                        float w = bucket_w[e * TMAX + m];
                        float* prow = part + (size_t)(base + m) * HDIM + h0;
                        prow[n]      = w * acc0[i][r];
                        prow[16 + n] = w * acc1[i][r];
                    }
                }
            }
        }
    } else {
        #pragma unroll
        for (int i = 0; i < 4; ++i) {
            #pragma unroll
            for (int r = 0; r < 8; ++r) {
                int m = m0 + 16 * i + r + 8 * half;
                if (m < cnt) {
                    int   t = bucket_tok[e * TMAX + m];
                    float w = bucket_w  [e * TMAX + m];
                    atomicAdd(&out[(size_t)t * HDIM + h0 + n],      w * acc0[i][r]);
                    atomicAdd(&out[(size_t)t * HDIM + h0 + 16 + n], w * acc1[i][r]);
                }
            }
        }
    }
}

// ---------------- combine: out[t,:] = shared + sum of 4 routed partials ----
__global__ __launch_bounds__(256) void combine_kernel(
    const float* __restrict__ part, const int* __restrict__ offsets,
    const int* __restrict__ tok_entry, float* __restrict__ out, int T)
{
    const int hv  = HDIM / 4;
    int idx = blockIdx.x * blockDim.x + threadIdx.x;
    int t   = idx / hv;
    int h4  = (idx % hv) * 4;
    if (t >= T) return;

    int sslot = offsets[NEXP] + t;                 // shared expert slot
    float4 s = *(const float4*)(part + (size_t)sslot * HDIM + h4);
    #pragma unroll
    for (int k = 0; k < KTOP; ++k) {
        int entry = tok_entry[t * KTOP + k];
        int slot  = offsets[entry >> 11] + (entry & (TMAX - 1));   // TMAX == 1<<11
        const float4 p = *(const float4*)(part + (size_t)slot * HDIM + h4);
        s.x += p.x; s.y += p.y; s.z += p.z; s.w += p.w;
    }
    *(float4*)(out + (size_t)t * HDIM + h4) = s;
}

// ---------------- host-side orchestration ----------------------------------
static inline void launch_cvt(const float* src, __bf16* dst, long n, hipStream_t s) {
    long blocks = (n / 8 + 255) / 256;
    cvt_bf16_kernel<<<(int)blocks, 256, 0, s>>>(src, dst, n);
}

extern "C" void kernel_launch(void* const* d_in, const int* in_sizes, int n_in,
                              void* d_out, int out_size, void* d_ws, size_t ws_size,
                              hipStream_t stream) {
    const float* x   = (const float*)d_in[0];   // [4,512,H]
    const float* rw  = (const float*)d_in[1];   // [E,H]
    const float* rb  = (const float*)d_in[2];   // [E]
    const float* eg  = (const float*)d_in[3];   // [E,F,H]
    const float* eu  = (const float*)d_in[4];   // [E,F,H]
    const float* ed  = (const float*)d_in[5];   // [E,H,F]
    const float* sg  = (const float*)d_in[6];   // [F,H]
    const float* su  = (const float*)d_in[7];   // [F,H]
    const float* sd  = (const float*)d_in[8];   // [H,F]
    float* out = (float*)d_out;

    int T = in_sizes[0] / HDIM;                 // 2048
    if (T > TMAX) T = TMAX;

    // ---- workspace carving (bytes) ----
    char*   ws      = (char*)d_ws;
    int*    counts  = (int*)(ws + 0);                        // 17 ints
    int*    offsets = (int*)(ws + 128);                      // 17 ints
    int*    btok    = (int*)(ws + 256);                      // 17*TMAX ints
    float*  bw      = (float*)(ws + 256 + NEXP1 * TMAX * 4);
    size_t  te_off  = 256 + (size_t)2 * NEXP1 * TMAX * 4;
    int*    tent    = (int*)(ws + te_off);                   // TMAX*KTOP ints
    size_t  actb_off = (te_off + (size_t)TMAX * KTOP * 4 + 255) & ~(size_t)255;
    __bf16* actb     = (__bf16*)(ws + actb_off);
    size_t  rows_max = (size_t)TMAX * KTOP + TMAX;           // 10240
    size_t  actb_end = actb_off + rows_max * FDIM * 2;       // ~20 MB

    size_t  part_off = (actb_end + 255) & ~(size_t)255;
    float*  part     = (float*)(ws + part_off);
    size_t  part_end = part_off + rows_max * HDIM * 4;       // ~80 MB

    // optional bf16 weight/x mirror (~222 MB)
    size_t nx  = (size_t)T * HDIM;
    size_t new_= (size_t)NEXP * FDIM * HDIM;
    size_t nsw = (size_t)FDIM * HDIM;
    size_t opt_off  = (part_end + 255) & ~(size_t)255;
    size_t opt_end  = opt_off + 2 * (nx + 3 * new_ + 3 * nsw);

    bool use_part = (ws_size >= part_end);
    bool use_bf16 = (ws_size >= opt_end);

    __bf16* xb  = (__bf16*)(ws + opt_off);
    __bf16* egb = xb  + nx;
    __bf16* eub = egb + new_;
    __bf16* edb = eub + new_;
    __bf16* sgb = edb + new_;
    __bf16* sub = sgb + nsw;
    __bf16* sdb = sub + nsw;

    int mtiles = (T + 63) / 64;

    moe_init_kernel<<<(out_size + 255) / 256, 256, 0, stream>>>(
        out, out_size, counts, btok, bw, T);

    router_kernel<<<(T + 3) / 4, 128, 0, stream>>>(
        x, rw, rb, counts, btok, bw, tent, T);

    offsets_kernel<<<1, 32, 0, stream>>>(counts, offsets);

    dim3 ggrid(FDIM / 64, mtiles, NEXP1);
    dim3 dgrid(HDIM / 128, mtiles, NEXP1);

    if (use_bf16) {
        launch_cvt(x,  xb,  (long)nx,  stream);
        launch_cvt(eg, egb, (long)new_, stream);
        launch_cvt(eu, eub, (long)new_, stream);
        launch_cvt(ed, edb, (long)new_, stream);
        launch_cvt(sg, sgb, (long)nsw, stream);
        launch_cvt(su, sub, (long)nsw, stream);
        launch_cvt(sd, sdb, (long)nsw, stream);

        gateup_kernel<__bf16><<<ggrid, 128, 0, stream>>>(
            xb, egb, eub, sgb, sub, counts, offsets, btok, actb);
        down_kernel<__bf16, true><<<dgrid, 128, 0, stream>>>(
            actb, edb, sdb, counts, offsets, btok, bw, part, out);
    } else {
        gateup_kernel<float><<<ggrid, 128, 0, stream>>>(
            x, eg, eu, sg, su, counts, offsets, btok, actb);
        if (use_part) {
            down_kernel<float, true><<<dgrid, 128, 0, stream>>>(
                actb, ed, sd, counts, offsets, btok, bw, part, out);
        } else {
            down_kernel<float, false><<<dgrid, 128, 0, stream>>>(
                actb, ed, sd, counts, offsets, btok, bw, part, out);
        }
    }

    if (use_part) {
        int total = T * (HDIM / 4);
        combine_kernel<<<(total + 255) / 256, 256, 0, stream>>>(
            part, offsets, tent, out, T);
    }
}
